// VectorQuantizer_21749714387651
// MI455X (gfx1250) — compile-verified, matrix operand order checked by reading, not run on hardware
//
#include <hip/hip_runtime.h>
#include <hip/hip_bf16.h>
#include <math.h>

typedef __attribute__((ext_vector_type(16))) _Float16 v16h;
typedef __attribute__((ext_vector_type(8)))  float    v8f;

#define N_LAT  32768   // 16*2048 latents
#define N_CODE 1024    // codebook entries
#define EDIM   64
#define SK_EPS 0.05f
#define SK_ITERS 100

// ---------- helpers ----------
__device__ __forceinline__ unsigned fkey(float f) {        // order-preserving f32->u32
  unsigned u = __float_as_uint(f);
  return (u & 0x80000000u) ? ~u : (u | 0x80000000u);
}
__device__ __forceinline__ float funkey(unsigned k) {
  unsigned u = (k & 0x80000000u) ? (k & 0x7FFFFFFFu) : ~k;
  return __uint_as_float(u);
}
__device__ __forceinline__ unsigned short f2bf(float f) {  // RNE f32->bf16
  unsigned u = __float_as_uint(f);
  u += 0x7FFFu + ((u >> 16) & 1u);
  return (unsigned short)(u >> 16);
}
__device__ __forceinline__ float bf_lo(unsigned u) { return __uint_as_float(u << 16); }
__device__ __forceinline__ float bf_hi(unsigned u) { return __uint_as_float(u & 0xFFFF0000u); }
__device__ __forceinline__ int finitef(float v) {
  return (__float_as_uint(v) & 0x7F800000u) != 0x7F800000u;
}

// ---------- init: row norms, col norms, sinkhorn state ----------
__global__ __launch_bounds__(256) void vq_init(
    const float* __restrict__ x, const float* __restrict__ W,
    float* __restrict__ xx, float* __restrict__ ww, float* __restrict__ t0,
    unsigned* dmax, unsigned* dmin, int* flag, float* lossacc) {
  int tid = blockIdx.x * 256 + threadIdx.x;
  if (tid < N_LAT) {
    const float4* p = (const float4*)(x + (size_t)tid * EDIM);
    float s = 0.f;
#pragma unroll
    for (int q = 0; q < 16; q++) { float4 v = p[q]; s += v.x*v.x + v.y*v.y + v.z*v.z + v.w*v.w; }
    xx[tid] = s;
  }
  if (tid < N_CODE) {
    const float4* p = (const float4*)(W + (size_t)tid * EDIM);
    float s = 0.f;
#pragma unroll
    for (int q = 0; q < 16; q++) { float4 v = p[q]; s += v.x*v.x + v.y*v.y + v.z*v.z + v.w*v.w; }
    ww[tid] = s;
    t0[tid] = 1.0f / (float)N_CODE;   // => c_0 == 1
  }
  if (tid == 0) { *dmax = 0u; *dmin = 0xFFFFFFFFu; *flag = 1; *lossacc = 0.f; }
}

// ---------- WMMA tile: two 16x16 tiles of x * W^T ----------
__device__ __forceinline__ void gemm_tile(const float* __restrict__ x,
                                          const float* __restrict__ W,
                                          int m0, int n0, int lane,
                                          v8f& acc0, v8f& acc1) {
  const int row  = m0 + (lane & 15);
  const int colA = n0 + (lane & 15);
  const int colB = colA + 16;
  const int ko   = (lane < 16) ? 0 : 8;   // A: lanes 16-31 hold K+8..; B symmetric
  v8f z = {0.f,0.f,0.f,0.f,0.f,0.f,0.f,0.f};
  acc0 = z; acc1 = z;
#pragma unroll
  for (int kc = 0; kc < EDIM; kc += 32) {
    v16h a, b0, b1;
    const float* xr = x + (size_t)row  * EDIM + kc + ko;
    const float* w0 = W + (size_t)colA * EDIM + kc + ko;
    const float* w1 = W + (size_t)colB * EDIM + kc + ko;
#pragma unroll
    for (int e = 0; e < 8; e++) {
      a[e]    = (_Float16)xr[e];      a[8+e]  = (_Float16)xr[16+e];
      b0[e]   = (_Float16)w0[e];      b0[8+e] = (_Float16)w0[16+e];
      b1[e]   = (_Float16)w1[e];      b1[8+e] = (_Float16)w1[16+e];
    }
    acc0 = __builtin_amdgcn_wmma_f32_16x16x32_f16(false, a, false, b0, (short)0, acc0, false, false);
    acc1 = __builtin_amdgcn_wmma_f32_16x16x32_f16(false, a, false, b1, (short)0, acc1, false, false);
  }
}

// ---------- pass 1: distances -> global max/min ----------
__global__ __launch_bounds__(256) void vq_dist_minmax(
    const float* __restrict__ x, const float* __restrict__ W,
    const float* __restrict__ xx, const float* __restrict__ ww,
    unsigned* dmax, unsigned* dmin) {
  const int wave = threadIdx.x >> 5, lane = threadIdx.x & 31;
  const int m0 = blockIdx.x * 64 + (wave & 3) * 16;
  const int n0 = blockIdx.y * 64 + (wave >> 2) * 32;
  v8f acc0, acc1;
  gemm_tile(x, W, m0, n0, lane, acc0, acc1);
  const int cr   = (lane < 16) ? 0 : 8;
  const int colA = n0 + (lane & 15);
  const float wwa = ww[colA], wwb = ww[colA + 16];
  float lmax = -3.4e38f, lmin = 3.4e38f;
#pragma unroll
  for (int v = 0; v < 8; v++) {
    const int rr = m0 + v + cr;
    const float xn = xx[rr];
    const float d0 = xn + wwa - 2.0f * acc0[v];
    const float d1 = xn + wwb - 2.0f * acc1[v];
    lmax = fmaxf(lmax, fmaxf(d0, d1));
    lmin = fminf(lmin, fminf(d0, d1));
  }
#pragma unroll
  for (int off = 16; off; off >>= 1) {
    lmax = fmaxf(lmax, __shfl_xor(lmax, off));
    lmin = fminf(lmin, __shfl_xor(lmin, off));
  }
  __shared__ float smax[8], smin[8];
  if (lane == 0) { smax[wave] = lmax; smin[wave] = lmin; }
  __syncthreads();
  if (threadIdx.x == 0) {
    float M = smax[0], m = smin[0];
    for (int i = 1; i < 8; i++) { M = fmaxf(M, smax[i]); m = fminf(m, smin[i]); }
    atomicMax(dmax, fkey(M));
    atomicMin(dmin, fkey(m));
  }
}

// ---------- pass 2: recompute d, A = exp(-(d-mid)/(amp*eps)) as bf16 ----------
__global__ __launch_bounds__(256) void vq_dist_exp(
    const float* __restrict__ x, const float* __restrict__ W,
    const float* __restrict__ xx, const float* __restrict__ ww,
    const unsigned* dmax, const unsigned* dmin,
    unsigned short* __restrict__ A) {
  const int wave = threadIdx.x >> 5, lane = threadIdx.x & 31;
  const int m0 = blockIdx.x * 64 + (wave & 3) * 16;
  const int n0 = blockIdx.y * 64 + (wave >> 2) * 32;
  v8f acc0, acc1;
  gemm_tile(x, W, m0, n0, lane, acc0, acc1);
  const float mx = funkey(*dmax), mn = funkey(*dmin);
  const float mid = 0.5f * (mx + mn);
  const float amp = mx - mid + 1e-5f;
  const float scale = -1.0f / (amp * SK_EPS);
  const int cr   = (lane < 16) ? 0 : 8;
  const int colA = n0 + (lane & 15);
  const float wwa = ww[colA], wwb = ww[colA + 16];
#pragma unroll
  for (int v = 0; v < 8; v++) {
    const int rr = m0 + v + cr;
    const float xn = xx[rr];
    const float d0 = xn + wwa - 2.0f * acc0[v];
    const float d1 = xn + wwb - 2.0f * acc1[v];
    A[(size_t)rr * N_CODE + colA]      = f2bf(expf((d0 - mid) * scale));
    A[(size_t)rr * N_CODE + colA + 16] = f2bf(expf((d1 - mid) * scale));
  }
}

// ---------- sinkhorn row pass: c = 1/(K*t_in); r = 1/(B * A c); zero t_out ----------
__global__ __launch_bounds__(256) void vq_row(
    const unsigned short* __restrict__ A, const float* __restrict__ t_in,
    float* __restrict__ r, float* __restrict__ t_out) {
  const int wave = threadIdx.x >> 5, lane = threadIdx.x & 31;
  if (blockIdx.x == 0) {
    for (int j = threadIdx.x; j < N_CODE; j += 256) t_out[j] = 0.f;
  }
  float c[32];
#pragma unroll
  for (int m = 0; m < 4; m++)
#pragma unroll
    for (int q = 0; q < 8; q++)
      c[m * 8 + q] = 1.0f / ((float)N_CODE * t_in[m * 256 + lane * 8 + q]);
  const int gw = blockIdx.x * 8 + wave;   // 4096 waves total
#pragma unroll 2
  for (int rr = 0; rr < 8; rr++) {
    const int row = gw * 8 + rr;
    const uint4* ap = (const uint4*)(A + (size_t)row * N_CODE);
    float s = 0.f;
#pragma unroll
    for (int m = 0; m < 4; m++) {
      uint4 u = ap[m * 32 + lane];        // 8 consecutive bf16, fully coalesced
      s += bf_lo(u.x) * c[m*8+0] + bf_hi(u.x) * c[m*8+1]
         + bf_lo(u.y) * c[m*8+2] + bf_hi(u.y) * c[m*8+3]
         + bf_lo(u.z) * c[m*8+4] + bf_hi(u.z) * c[m*8+5]
         + bf_lo(u.w) * c[m*8+6] + bf_hi(u.w) * c[m*8+7];
    }
#pragma unroll
    for (int off = 16; off; off >>= 1) s += __shfl_xor(s, off);
    if (lane == 0) r[row] = 1.0f / ((float)N_LAT * s);
  }
}

// ---------- sinkhorn col pass: t_out[j] += sum_i r[i]*A[i][j] ----------
// Each thread owns two adjacent columns via one u32 (2x bf16) load per row step.
__global__ __launch_bounds__(256) void vq_col(
    const unsigned short* __restrict__ A, const float* __restrict__ r,
    float* __restrict__ t_out) {
  const int cb = blockIdx.x & 1;          // 2 column blocks of 512 columns
  const int rb = blockIdx.x >> 1;         // 128 row stripes of 256 rows
  __shared__ float rs[256];
  const int i0 = rb * 256;
  rs[threadIdx.x] = r[i0 + threadIdx.x];
  __syncthreads();
  const int j = cb * 512 + threadIdx.x * 2;          // columns j, j+1
  const unsigned* ap = (const unsigned*)(A + (size_t)i0 * N_CODE + j);
  float acc0 = 0.f, acc1 = 0.f;
#pragma unroll 8
  for (int i = 0; i < 256; i++) {
    const unsigned u = ap[(size_t)i * (N_CODE / 2)];  // u32 stride = row stride in pairs
    const float rv = rs[i];
    acc0 += rv * bf_lo(u);
    acc1 += rv * bf_hi(u);
  }
  atomicAdd(&t_out[j],     acc0);
  atomicAdd(&t_out[j + 1], acc1);
}

// ---------- finalize: c from t, finiteness flag ----------
__global__ __launch_bounds__(256) void vq_finalize(
    const float* __restrict__ r, const float* __restrict__ t_in,
    float* __restrict__ c, int* flag) {
  int tid = blockIdx.x * 256 + threadIdx.x;
  if (tid < N_LAT) { if (!finitef(r[tid])) *flag = 0; }
  if (tid < N_CODE) {
    float cc = 1.0f / ((float)N_CODE * t_in[tid]);
    c[tid] = cc;
    if (!finitef(cc)) *flag = 0;
  }
}

// ---------- argmax_j A[i,j]*c'[j]; gather W row; loss partials ----------
__global__ __launch_bounds__(256) void vq_argmax_out(
    const unsigned short* __restrict__ A, const float* __restrict__ c,
    const int* __restrict__ flag, const float* __restrict__ W,
    const float* __restrict__ x, float* __restrict__ out,
    float* __restrict__ lossacc) {
  const int wave = threadIdx.x >> 5, lane = threadIdx.x & 31;
  const int row = blockIdx.x * 8 + wave;
  const int bad = (*flag == 0);           // fallback argmin d == argmax A (c'=1)
  float best = -3.4e38f; int bi = 0;
  const uint4* ap = (const uint4*)(A + (size_t)row * N_CODE);
#pragma unroll
  for (int m = 0; m < 4; m++) {
    uint4 u = ap[m * 32 + lane];
    const unsigned us[4] = {u.x, u.y, u.z, u.w};
#pragma unroll
    for (int q = 0; q < 4; q++) {
      const int col = m * 256 + lane * 8 + q * 2;
      const float v0 = bf_lo(us[q]) * (bad ? 1.0f : c[col]);
      const float v1 = bf_hi(us[q]) * (bad ? 1.0f : c[col + 1]);
      if (v0 > best) { best = v0; bi = col; }
      if (v1 > best) { best = v1; bi = col + 1; }
    }
  }
#pragma unroll
  for (int off = 16; off; off >>= 1) {
    float ob = __shfl_xor(best, off);
    int   oi = __shfl_xor(bi, off);
    if (ob > best || (ob == best && oi < bi)) { best = ob; bi = oi; }
  }
  const int j = bi;                        // identical on all lanes after xor-reduce
  const float w0 = W[(size_t)j * EDIM + lane * 2];
  const float w1 = W[(size_t)j * EDIM + lane * 2 + 1];
  const float x0 = x[(size_t)row * EDIM + lane * 2];
  const float x1 = x[(size_t)row * EDIM + lane * 2 + 1];
  out[(size_t)row * EDIM + lane * 2]     = w0;   // x_q_st value == x_q
  out[(size_t)row * EDIM + lane * 2 + 1] = w1;
  float ds = (w0 - x0) * (w0 - x0) + (w1 - x1) * (w1 - x1);
#pragma unroll
  for (int off = 16; off; off >>= 1) ds += __shfl_xor(ds, off);
  if (lane == 0) {
    atomicAdd(lossacc, ds);
    out[(size_t)N_LAT * EDIM + 1 + row] = (float)j;   // indices output
  }
}

__global__ void vq_loss(const float* lossacc, float* out) {
  // codebook_loss + beta*commitment_loss; both terms share the same value => (1+beta)*mse
  out[(size_t)N_LAT * EDIM] = 1.25f * (*lossacc) * (1.0f / ((float)N_LAT * (float)EDIM));
}

extern "C" void kernel_launch(void* const* d_in, const int* in_sizes, int n_in,
                              void* d_out, int out_size, void* d_ws, size_t ws_size,
                              hipStream_t stream) {
  (void)in_sizes; (void)n_in; (void)out_size; (void)ws_size;
  const float* x = (const float*)d_in[0];
  const float* W = (const float*)d_in[1];
  float* out = (float*)d_out;
  char* ws = (char*)d_ws;

  size_t off = 0;
  unsigned short* A = (unsigned short*)(ws + off); off += (size_t)N_LAT * N_CODE * 2;  // 67 MB, L2-resident
  float* r  = (float*)(ws + off); off += (size_t)N_LAT * 4;
  float* t0 = (float*)(ws + off); off += (size_t)N_CODE * 4;
  float* t1 = (float*)(ws + off); off += (size_t)N_CODE * 4;
  float* c  = (float*)(ws + off); off += (size_t)N_CODE * 4;
  float* xx = (float*)(ws + off); off += (size_t)N_LAT * 4;
  float* ww = (float*)(ws + off); off += (size_t)N_CODE * 4;
  unsigned* dmax = (unsigned*)(ws + off); off += 16;
  unsigned* dmin = dmax + 1;
  int*      flag = (int*)(dmax + 2);
  float* lossacc = (float*)(dmax + 3);

  vq_init<<<N_LAT / 256, 256, 0, stream>>>(x, W, xx, ww, t0, dmax, dmin, flag, lossacc);
  vq_dist_minmax<<<dim3(N_LAT / 64, N_CODE / 64), 256, 0, stream>>>(x, W, xx, ww, dmax, dmin);
  vq_dist_exp<<<dim3(N_LAT / 64, N_CODE / 64), 256, 0, stream>>>(x, W, xx, ww, dmax, dmin, A);

  for (int it = 0; it < SK_ITERS; ++it) {
    float* tin  = (it & 1) ? t1 : t0;
    float* tout = (it & 1) ? t0 : t1;
    vq_row<<<512, 256, 0, stream>>>(A, tin, r, tout);
    vq_col<<<256, 256, 0, stream>>>(A, r, tout);
  }
  // SK_ITERS even: final column sums ended in t0
  vq_finalize<<<N_LAT / 256, 256, 0, stream>>>(r, t0, c, flag);
  vq_argmax_out<<<N_LAT / 8 / 32, 256, 0, stream>>>(A, c, flag, W, x, out, lossacc);
  vq_loss<<<1, 1, 0, stream>>>(lossacc, out);
}
